// MambaFlowModel_82755429859945
// MI455X (gfx1250) — matrix operand order repaired and checked by reading, not compiled, not run on hardware
//
#include <hip/hip_runtime.h>

// ---------------------------------------------------------------------------
// MambaFlowModel forward for MI455X (gfx1250), wave32, WMMA bf16 split-precision
// ---------------------------------------------------------------------------

#define D_MODEL   512
#define D_STATE   16
#define D_CONV    4
#define D_INNER   1024
#define DT_RANK   32
#define N_LAYERS  6
#define VOCAB     8192
#define BATCH     2
#define SEQ       512
#define MROWS     (BATCH * SEQ)      // 1024
#define COND_DIM  224
#define FLOW_HID  512
#define EXPR_IN   4
#define EXPR_HID  256
#define EXPR_EMB  128

typedef __attribute__((ext_vector_type(16))) __bf16 v16bf;
typedef __attribute__((ext_vector_type(8)))  float  v8f;

// ---------------- small device helpers ----------------
__device__ __forceinline__ float gelu_f(float x) {           // exact erf GELU
  return 0.5f * x * (1.0f + erff(x * 0.70710678118654752f));
}
__device__ __forceinline__ float silu_f(float x) {
  return x / (1.0f + expf(-x));
}
__device__ __forceinline__ float softplus_f(float x) {
  return (x > 20.0f) ? x : log1pf(expf(x));
}

// ---------------------------------------------------------------------------
// One 32-wide K step of the split-precision bf16 WMMA GEMM.
// NG: need per-lane column guard (partial N tile). KG: need K guard (K tail).
// Fast path (!KG): fully unconditional float4 (global_load_b128) loads.
// ---------------------------------------------------------------------------
template<bool NG, bool KG>
__device__ __forceinline__ void ktile(
    const float* __restrict__ Arow, const float* __restrict__ Wrow,
    int k0, int K, int half, bool colok, v8f& acc)
{
  float af[16], bf[16];
  if (!KG) {
    // A: lane reads A[row, k0+8h .. +7] and A[row, k0+16+8h .. +7]
    const float4 a0 = *(const float4*)(Arow + k0 + half * 8);
    const float4 a1 = *(const float4*)(Arow + k0 + half * 8 + 4);
    const float4 a2 = *(const float4*)(Arow + k0 + 16 + half * 8);
    const float4 a3 = *(const float4*)(Arow + k0 + 16 + half * 8 + 4);
    af[0]=a0.x; af[1]=a0.y; af[2]=a0.z;  af[3]=a0.w;
    af[4]=a1.x; af[5]=a1.y; af[6]=a1.z;  af[7]=a1.w;
    af[8]=a2.x; af[9]=a2.y; af[10]=a2.z; af[11]=a2.w;
    af[12]=a3.x; af[13]=a3.y; af[14]=a3.z; af[15]=a3.w;
    if (!NG || colok) {
      // B: lane reads W[col, k0+8h .. +7] and W[col, k0+16+8h .. +7]
      const float4 b0 = *(const float4*)(Wrow + k0 + half * 8);
      const float4 b1 = *(const float4*)(Wrow + k0 + half * 8 + 4);
      const float4 b2 = *(const float4*)(Wrow + k0 + 16 + half * 8);
      const float4 b3 = *(const float4*)(Wrow + k0 + 16 + half * 8 + 4);
      bf[0]=b0.x; bf[1]=b0.y; bf[2]=b0.z;  bf[3]=b0.w;
      bf[4]=b1.x; bf[5]=b1.y; bf[6]=b1.z;  bf[7]=b1.w;
      bf[8]=b2.x; bf[9]=b2.y; bf[10]=b2.z; bf[11]=b2.w;
      bf[12]=b3.x; bf[13]=b3.y; bf[14]=b3.z; bf[15]=b3.w;
    } else {
#pragma unroll
      for (int e = 0; e < 16; ++e) bf[e] = 0.f;
    }
  } else {
#pragma unroll
    for (int e = 0; e < 16; ++e) {
      int ka = k0 + ((e < 8) ? (half * 8 + e) : (16 + half * 8 + (e - 8)));
      af[e] = (ka < K) ? Arow[ka] : 0.f;
      int kb = ka;
      bf[e] = ((!NG || colok) && kb < K) ? Wrow[kb] : 0.f;
    }
  }
  v16bf ahi, alo, bhi, blo;
#pragma unroll
  for (int e = 0; e < 16; ++e) {
    __bf16 ah = (__bf16)af[e];            // hw RNE cvt
    ahi[e] = ah;
    alo[e] = (__bf16)(af[e] - (float)ah); // residual, ~8 extra mantissa bits
    __bf16 bh = (__bf16)bf[e];
    bhi[e] = bh;
    blo[e] = (__bf16)(bf[e] - (float)bh);
  }
  acc = __builtin_amdgcn_wmma_f32_16x16x32_bf16(false, ahi, false, bhi, (short)0, acc, false, false);
  acc = __builtin_amdgcn_wmma_f32_16x16x32_bf16(false, alo, false, bhi, (short)0, acc, false, false);
  acc = __builtin_amdgcn_wmma_f32_16x16x32_bf16(false, ahi, false, blo, (short)0, acc, false, false);
}

// ---------------------------------------------------------------------------
// Generic GEMM: C[M,N] = A[M,K] * W[N,K]^T (+bias[N]) (+= existing C if acc)
// Requires M % 16 == 0 (always 1024 here). One wave per 16x16 C tile.
// ---------------------------------------------------------------------------
__global__ __launch_bounds__(256) void gemm_wmma_kernel(
    const float* __restrict__ A, const float* __restrict__ W,
    float* __restrict__ C, const float* __restrict__ bias,
    int M, int N, int K, int lda, int accumulate)
{
  const int lane = threadIdx.x & 31;
  const int wave = threadIdx.x >> 5;
  const int tiles_n = (N + 15) >> 4;
  const int tiles_m = M >> 4;
  const int tile = blockIdx.x * 8 + wave;          // wave-uniform
  if (tile >= tiles_m * tiles_n) return;           // uniform exit, EXEC stays full

  const int tm   = tile / tiles_n;
  const int tn   = tile % tiles_n;
  const int sub  = lane & 15;
  const int half = lane >> 4;
  const int row  = tm * 16 + sub;                  // A row handled by this lane
  const int col  = tn * 16 + sub;                  // B/C column handled by this lane

  v8f acc = {0.f, 0.f, 0.f, 0.f, 0.f, 0.f, 0.f, 0.f};
  const bool colok = (col < N);
  const bool fulln = (tn * 16 + 16 <= N);          // wave-uniform
  const float* Arow = A + (long)row * lda;
  const float* Wrow = W + (long)col * K;
  const int kmain = K & ~31;

  if (fulln) {
    for (int k0 = 0; k0 < kmain; k0 += 32)
      ktile<false, false>(Arow, Wrow, k0, K, half, true, acc);
    if (kmain < K)
      ktile<false, true>(Arow, Wrow, kmain, K, half, true, acc);
  } else {
    for (int k0 = 0; k0 < kmain; k0 += 32)
      ktile<true, false>(Arow, Wrow, k0, K, half, colok, acc);
    if (kmain < K)
      ktile<true, true>(Arow, Wrow, kmain, K, half, colok, acc);
  }

  const float bv = (bias != nullptr && colok) ? bias[col] : 0.f;
#pragma unroll
  for (int i = 0; i < 8; ++i) {
    int m = tm * 16 + half * 8 + i;                // C/D layout: VGPR i -> row 8*half+i
    if (colok) {
      float r = acc[i] + bv;
      if (accumulate) r += C[(long)m * N + col];
      C[(long)m * N + col] = r;
    }
  }
}

// ---------------------------------------------------------------------------
// LayerNorm over last dim W (<=512), one block per row; optional exact-GELU
// applied to the input before normalization (for _ln(_gelu(x))).
// Safe in-place (reads staged to LDS before writes).
// ---------------------------------------------------------------------------
__global__ __launch_bounds__(256) void ln_kernel(
    const float* __restrict__ in, float* __restrict__ out,
    const float* __restrict__ g, const float* __restrict__ b,
    int W, int apply_gelu)
{
  __shared__ float vals[512];
  __shared__ float red[256];
  const int row = blockIdx.x;
  const float* x = in + (long)row * W;
  float s = 0.f;
  for (int j = threadIdx.x; j < W; j += 256) {
    float v = x[j];
    if (apply_gelu) v = gelu_f(v);
    vals[j] = v;
    s += v;
  }
  red[threadIdx.x] = s; __syncthreads();
  for (int o = 128; o > 0; o >>= 1) { if (threadIdx.x < o) red[threadIdx.x] += red[threadIdx.x + o]; __syncthreads(); }
  const float mu = red[0] / (float)W; __syncthreads();
  float s2 = 0.f;
  for (int j = threadIdx.x; j < W; j += 256) { float d = vals[j] - mu; s2 += d * d; }
  red[threadIdx.x] = s2; __syncthreads();
  for (int o = 128; o > 0; o >>= 1) { if (threadIdx.x < o) red[threadIdx.x] += red[threadIdx.x + o]; __syncthreads(); }
  const float rstd = rsqrtf(red[0] / (float)W + 1e-5f); __syncthreads();
  for (int j = threadIdx.x; j < W; j += 256)
    out[(long)row * W + j] = (vals[j] - mu) * rstd * g[j] + b[j];
}

// ---------------- token embedding ----------------
__global__ __launch_bounds__(256) void embed_kernel(
    const int* __restrict__ tokens, const float* __restrict__ emb, float* __restrict__ x)
{
  int idx = blockIdx.x * 256 + threadIdx.x;            // MROWS * D_MODEL
  if (idx >= MROWS * D_MODEL) return;
  int m = idx >> 9, j = idx & 511;
  x[idx] = emb[(long)tokens[m] * D_MODEL + j];
}

// ---------------- expression encoder stage 1: gelu(expr @ e1_W^T + b), K=4 ----
__global__ __launch_bounds__(256) void expr1_kernel(
    const float* __restrict__ expr, const float* __restrict__ w,
    const float* __restrict__ b, float* __restrict__ out)
{
  int idx = blockIdx.x * 256 + threadIdx.x;            // MROWS * EXPR_HID
  if (idx >= MROWS * EXPR_HID) return;
  int m = idx >> 8, j = idx & 255;
  float s = b[j];
#pragma unroll
  for (int i = 0; i < EXPR_IN; ++i) s += expr[m * EXPR_IN + i] * w[j * EXPR_IN + i];
  out[idx] = gelu_f(s);
}

// ---------------- conditioning module (per batch) ----------------
__global__ __launch_bounds__(256) void cond_kernel(
    const int* __restrict__ mood, const int* __restrict__ raga, const int* __restrict__ taal,
    const float* __restrict__ tempo, const float* __restrict__ dur,
    const float* __restrict__ mood_emb, const float* __restrict__ raga_emb,
    const float* __restrict__ taal_emb,
    const float* __restrict__ tempo_W, const float* __restrict__ tempo_b,
    const float* __restrict__ dur_W, const float* __restrict__ dur_b,
    const float* __restrict__ cproj_W, const float* __restrict__ cproj_b,
    const float* __restrict__ cg, const float* __restrict__ cb,
    float* __restrict__ cond)
{
  __shared__ float vec[COND_DIM];
  __shared__ float vals[D_MODEL];
  __shared__ float red[256];
  const int b = blockIdx.x, t = threadIdx.x;
  if      (t <  64) vec[t] = mood_emb[mood[b] * 64 + t];
  else if (t < 128) vec[t] = raga_emb[raga[b] * 64 + (t - 64)];
  else if (t < 160) vec[t] = taal_emb[taal[b] * 32 + (t - 128)];
  else if (t < 192) vec[t] = tempo[b] * tempo_W[t - 160] + tempo_b[t - 160];
  else if (t < 224) vec[t] = dur[b] * dur_W[t - 192] + dur_b[t - 192];
  __syncthreads();
  for (int j = t; j < D_MODEL; j += 256) {
    float s = cproj_b[j];
    const float* wr = cproj_W + (long)j * COND_DIM;
    for (int i = 0; i < COND_DIM; ++i) s += vec[i] * wr[i];
    vals[j] = gelu_f(s);
  }
  __syncthreads();
  float s = 0.f;
  for (int j = t; j < D_MODEL; j += 256) s += vals[j];
  red[t] = s; __syncthreads();
  for (int o = 128; o > 0; o >>= 1) { if (t < o) red[t] += red[t + o]; __syncthreads(); }
  const float mu = red[0] / (float)D_MODEL; __syncthreads();
  float s2 = 0.f;
  for (int j = t; j < D_MODEL; j += 256) { float d = vals[j] - mu; s2 += d * d; }
  red[t] = s2; __syncthreads();
  for (int o = 128; o > 0; o >>= 1) { if (t < o) red[t] += red[t + o]; __syncthreads(); }
  const float rstd = rsqrtf(red[0] / (float)D_MODEL + 1e-5f); __syncthreads();
  for (int j = t; j < D_MODEL; j += 256)
    cond[b * D_MODEL + j] = (vals[j] - mu) * rstd * cg[j] + cb[j];
}

// ---------------- x += cond + taal positional embeddings ----------------
__global__ __launch_bounds__(256) void addmisc_kernel(
    float* __restrict__ x, const float* __restrict__ cond,
    const float* __restrict__ cyc_emb, const float* __restrict__ str_emb,
    const int* __restrict__ cyc_len)
{
  int idx = blockIdx.x * 256 + threadIdx.x;
  if (idx >= MROWS * D_MODEL) return;
  int m = idx >> 9, j = idx & 511;
  int b = m >> 9, l = m & 511;
  int cl  = *cyc_len;
  int pm  = (cl < 16) ? cl : 16;
  int pos = l % pm;
  int st  = (l % cl == 0) ? 0 : 3;
  x[idx] += cond[b * D_MODEL + j] + cyc_emb[pos * D_MODEL + j] + str_emb[st * D_MODEL + j];
}

// ---------------- depthwise causal conv(k=4) + SiLU ----------------
__global__ __launch_bounds__(256) void conv_kernel(
    const float* __restrict__ xz, const float* __restrict__ w,
    const float* __restrict__ bias, float* __restrict__ xc)
{
  int idx = blockIdx.x * 256 + threadIdx.x;            // MROWS * D_INNER
  if (idx >= MROWS * D_INNER) return;
  int m = idx >> 10, d = idx & 1023;
  int b = m >> 9, l = m & 511;
  float s = bias[d];
#pragma unroll
  for (int k = 0; k < D_CONV; ++k) {
    int ll = l - (D_CONV - 1) + k;
    if (ll >= 0) s += xz[((long)((b << 9) + ll)) * (2 * D_INNER) + d] * w[d * D_CONV + k];
  }
  xc[idx] = silu_f(s);
}

// ---------------- softplus in place ----------------
__global__ __launch_bounds__(256) void softplus_kernel(float* __restrict__ p, int n)
{
  int idx = blockIdx.x * 256 + threadIdx.x;
  if (idx < n) p[idx] = softplus_f(p[idx]);
}

// ---------------- selective scan: one thread per (b, d), 16 states in regs ----
__global__ __launch_bounds__(256) void scan_kernel(
    const float* __restrict__ dtb, const float* __restrict__ xc,
    const float* __restrict__ sp, const float* __restrict__ A_log,
    float* __restrict__ ys)
{
  int idx = blockIdx.x * 256 + threadIdx.x;
  if (idx >= BATCH * D_INNER) return;
  const int b = idx >> 10, d = idx & 1023;
  float A[D_STATE], h[D_STATE];
#pragma unroll
  for (int n = 0; n < D_STATE; ++n) { A[n] = -expf(A_log[d * D_STATE + n]); h[n] = 0.f; }
  for (int l = 0; l < SEQ; ++l) {
    const long m = (long)(b * SEQ + l);
    const float dt = dtb[m * D_INNER + d];
    const float xv = xc [m * D_INNER + d];
    const float dx = dt * xv;
    const float* bc = sp + m * (DT_RANK + 2 * D_STATE);
    float acc = 0.f;
#pragma unroll
    for (int n = 0; n < D_STATE; ++n) {
      float e = expf(dt * A[n]);
      h[n] = e * h[n] + dx * bc[DT_RANK + n];
      acc += h[n] * bc[DT_RANK + D_STATE + n];
    }
    ys[m * D_INNER + d] = acc;
  }
}

// ---------------- y = (ys + xc*D) * silu(z) ----------------
__global__ __launch_bounds__(256) void gate_kernel(
    const float* __restrict__ ys, const float* __restrict__ xc,
    const float* __restrict__ xz, const float* __restrict__ Dp,
    float* __restrict__ y)
{
  int idx = blockIdx.x * 256 + threadIdx.x;
  if (idx >= MROWS * D_INNER) return;
  int m = idx >> 10, d = idx & 1023;
  float z = xz[(long)m * (2 * D_INNER) + D_INNER + d];
  y[idx] = (ys[idx] + xc[idx] * Dp[d]) * silu_f(z);
}

// ---------------- flow time embedding (per batch) ----------------
__global__ __launch_bounds__(256) void temb_kernel(
    const float* __restrict__ ts,
    const float* __restrict__ w1, const float* __restrict__ b1,
    const float* __restrict__ w2, const float* __restrict__ b2,
    float* __restrict__ temb)
{
  __shared__ float h1[FLOW_HID];
  const int b = blockIdx.x;
  const float t = ts[b];
  for (int j = threadIdx.x; j < FLOW_HID; j += 256) h1[j] = silu_f(t * w1[j] + b1[j]);
  __syncthreads();
  for (int j = threadIdx.x; j < D_MODEL; j += 256) {
    float s = b2[j];
    const float* wr = w2 + (long)j * FLOW_HID;
    for (int i = 0; i < FLOW_HID; ++i) s += h1[i] * wr[i];
    temb[b * D_MODEL + j] = s;
  }
}

// ---------------- concat [hidden + temb, noisy_expression] -> [M, 516] -------
__global__ __launch_bounds__(256) void cat_kernel(
    const float* __restrict__ hid, const float* __restrict__ temb,
    const float* __restrict__ noisy, float* __restrict__ cat)
{
  const int W = D_MODEL + EXPR_IN;                     // 516
  int idx = blockIdx.x * 256 + threadIdx.x;
  if (idx >= MROWS * W) return;
  int m = idx / W, j = idx % W;
  int b = m >> 9;
  cat[idx] = (j < D_MODEL) ? (hid[(long)m * D_MODEL + j] + temb[b * D_MODEL + j])
                           : noisy[m * EXPR_IN + (j - D_MODEL)];
}

// ---------------------------------------------------------------------------
// Host orchestration
// ---------------------------------------------------------------------------
extern "C" void kernel_launch(void* const* d_in, const int* in_sizes, int n_in,
                              void* d_out, int out_size, void* d_ws, size_t ws_size,
                              hipStream_t stream)
{
  (void)in_sizes; (void)n_in; (void)out_size; (void)ws_size;
  // Input flattening: setup_inputs() insertion order, nested dicts depth-first.
  const int*   tokens   = (const int*)  d_in[0];
  const int*   mood     = (const int*)  d_in[1];
  const int*   raga     = (const int*)  d_in[2];
  const int*   taal     = (const int*)  d_in[3];
  const float* tempo    = (const float*)d_in[4];
  const float* duration = (const float*)d_in[5];
  const float* expr     = (const float*)d_in[6];
  const float* flow_t   = (const float*)d_in[7];
  const float* noisy    = (const float*)d_in[8];
  const int*   cyc_len  = (const int*)  d_in[9];
  // params
  const float* tok_emb  = (const float*)d_in[10];
  const float* mood_emb = (const float*)d_in[11];
  const float* raga_emb = (const float*)d_in[12];
  const float* taal_emb = (const float*)d_in[13];
  const float* tempo_W  = (const float*)d_in[14];
  const float* tempo_b  = (const float*)d_in[15];
  const float* dur_W    = (const float*)d_in[16];
  const float* dur_b    = (const float*)d_in[17];
  const float* cproj_W  = (const float*)d_in[18];
  const float* cproj_b  = (const float*)d_in[19];
  const float* cln_g    = (const float*)d_in[20];
  const float* cln_b    = (const float*)d_in[21];
  const float* e1_W     = (const float*)d_in[22];
  const float* e1_b     = (const float*)d_in[23];
  const float* eln1_g   = (const float*)d_in[24];
  const float* eln1_b   = (const float*)d_in[25];
  const float* e2_W     = (const float*)d_in[26];
  const float* e2_b     = (const float*)d_in[27];
  const float* eln2_g   = (const float*)d_in[28];
  const float* eln2_b   = (const float*)d_in[29];
  // 30 attn_W, 31 attn_b: dead code in reference (global_emb unused)
  const float* eproj_W  = (const float*)d_in[32];
  const float* eproj_b  = (const float*)d_in[33];
  const float* cycle_emb    = (const float*)d_in[34];
  const float* strength_emb = (const float*)d_in[35];
  const int LBASE = 36;                     // 11 tensors per layer
  const float* fln_g  = (const float*)d_in[LBASE + 11 * N_LAYERS + 0];
  const float* fln_b  = (const float*)d_in[LBASE + 11 * N_LAYERS + 1];
  const float* th_W   = (const float*)d_in[LBASE + 11 * N_LAYERS + 2];
  const float* th_b   = (const float*)d_in[LBASE + 11 * N_LAYERS + 3];
  const float* eh_W   = (const float*)d_in[LBASE + 11 * N_LAYERS + 4];
  const float* eh_b   = (const float*)d_in[LBASE + 11 * N_LAYERS + 5];
  const float* ft1_W  = (const float*)d_in[LBASE + 11 * N_LAYERS + 6];
  const float* ft1_b  = (const float*)d_in[LBASE + 11 * N_LAYERS + 7];
  const float* ft2_W  = (const float*)d_in[LBASE + 11 * N_LAYERS + 8];
  const float* ft2_b  = (const float*)d_in[LBASE + 11 * N_LAYERS + 9];
  const float* fv1_W  = (const float*)d_in[LBASE + 11 * N_LAYERS + 10];
  const float* fv1_b  = (const float*)d_in[LBASE + 11 * N_LAYERS + 11];
  const float* fvln_g = (const float*)d_in[LBASE + 11 * N_LAYERS + 12];
  const float* fvln_b = (const float*)d_in[LBASE + 11 * N_LAYERS + 13];
  const float* fv2_W  = (const float*)d_in[LBASE + 11 * N_LAYERS + 14];
  const float* fv2_b  = (const float*)d_in[LBASE + 11 * N_LAYERS + 15];

  // workspace layout (floats) — all offsets multiples of 16 floats (64B)
  float* ws   = (float*)d_ws;
  float* x    = ws;                         // [1024,512]
  float* fe1  = x    + MROWS * D_MODEL;     // [1024,256]
  float* fe   = fe1  + MROWS * EXPR_HID;    // [1024,128]
  float* cond = fe   + MROWS * EXPR_EMB;    // [2,512]
  float* xn   = cond + BATCH * D_MODEL;     // [1024,512]
  float* xz   = xn   + MROWS * D_MODEL;     // [1024,2048]
  float* xc   = xz   + MROWS * 2 * D_INNER; // [1024,1024]
  float* sp   = xc   + MROWS * D_INNER;     // [1024,64]
  float* dtb  = sp   + MROWS * (DT_RANK + 2 * D_STATE);  // [1024,1024]
  float* ysb  = dtb  + MROWS * D_INNER;     // [1024,1024]
  float* yb   = ysb  + MROWS * D_INNER;     // [1024,1024]
  float* hid  = yb   + MROWS * D_INNER;     // [1024,512]
  float* temb = hid  + MROWS * D_MODEL;     // [2,512]
  float* cat  = temb + BATCH * D_MODEL;     // [1024,516]
  float* vtmp = cat  + MROWS * (D_MODEL + EXPR_IN);      // [1024,512]

  float* out_logits = (float*)d_out;                     // [1024,8192]
  float* out_pred   = out_logits + (long)MROWS * VOCAB;  // [1024,4]
  float* out_v      = out_pred   + MROWS * EXPR_IN;      // [1024,4]

  auto gemm = [&](const float* A, const float* W, float* C, const float* bias,
                  int M, int N, int K, int lda, int acc) {
    int tiles = (M / 16) * ((N + 15) / 16);
    gemm_wmma_kernel<<<(tiles + 7) / 8, 256, 0, stream>>>(A, W, C, bias, M, N, K, lda, acc);
  };
  const int EW = 256;

  // --- embedding + expression encoder ---
  embed_kernel<<<(MROWS * D_MODEL + EW - 1) / EW, EW, 0, stream>>>(tokens, tok_emb, x);
  expr1_kernel<<<(MROWS * EXPR_HID + EW - 1) / EW, EW, 0, stream>>>(expr, e1_W, e1_b, fe1);
  ln_kernel<<<MROWS, EW, 0, stream>>>(fe1, fe1, eln1_g, eln1_b, EXPR_HID, 0);
  gemm(fe1, e2_W, fe, e2_b, MROWS, EXPR_EMB, EXPR_HID, EXPR_HID, 0);
  ln_kernel<<<MROWS, EW, 0, stream>>>(fe, fe, eln2_g, eln2_b, EXPR_EMB, 0);
  gemm(fe, eproj_W, x, eproj_b, MROWS, D_MODEL, EXPR_EMB, EXPR_EMB, /*acc=*/1);

  // --- conditioning + taal positions ---
  cond_kernel<<<BATCH, EW, 0, stream>>>(mood, raga, taal, tempo, duration,
      mood_emb, raga_emb, taal_emb, tempo_W, tempo_b, dur_W, dur_b,
      cproj_W, cproj_b, cln_g, cln_b, cond);
  addmisc_kernel<<<(MROWS * D_MODEL + EW - 1) / EW, EW, 0, stream>>>(
      x, cond, cycle_emb, strength_emb, cyc_len);

  // --- Mamba layers ---
  for (int l = 0; l < N_LAYERS; ++l) {
    const int base = LBASE + 11 * l;
    const float* ng     = (const float*)d_in[base + 0];
    const float* nb     = (const float*)d_in[base + 1];
    const float* in_W   = (const float*)d_in[base + 2];
    const float* conv_W = (const float*)d_in[base + 3];
    const float* conv_b = (const float*)d_in[base + 4];
    const float* x_W    = (const float*)d_in[base + 5];
    const float* dt_W   = (const float*)d_in[base + 6];
    const float* dt_b   = (const float*)d_in[base + 7];
    const float* A_log  = (const float*)d_in[base + 8];
    const float* Dp     = (const float*)d_in[base + 9];
    const float* out_W  = (const float*)d_in[base + 10];

    ln_kernel<<<MROWS, EW, 0, stream>>>(x, xn, ng, nb, D_MODEL, 0);
    gemm(xn, in_W, xz, nullptr, MROWS, 2 * D_INNER, D_MODEL, D_MODEL, 0);
    conv_kernel<<<(MROWS * D_INNER + EW - 1) / EW, EW, 0, stream>>>(xz, conv_W, conv_b, xc);
    gemm(xc, x_W, sp, nullptr, MROWS, DT_RANK + 2 * D_STATE, D_INNER, D_INNER, 0);
    gemm(sp, dt_W, dtb, dt_b, MROWS, D_INNER, DT_RANK, DT_RANK + 2 * D_STATE, 0);
    softplus_kernel<<<(MROWS * D_INNER + EW - 1) / EW, EW, 0, stream>>>(dtb, MROWS * D_INNER);
    scan_kernel<<<(BATCH * D_INNER + EW - 1) / EW, EW, 0, stream>>>(dtb, xc, sp, A_log, ysb);
    gate_kernel<<<(MROWS * D_INNER + EW - 1) / EW, EW, 0, stream>>>(ysb, xc, xz, Dp, yb);
    gemm(yb, out_W, x, nullptr, MROWS, D_MODEL, D_INNER, D_INNER, /*acc=*/1);
  }

  // --- heads ---
  ln_kernel<<<MROWS, EW, 0, stream>>>(x, hid, fln_g, fln_b, D_MODEL, 0);
  gemm(hid, th_W, out_logits, th_b, MROWS, VOCAB, D_MODEL, D_MODEL, 0);
  gemm(hid, eh_W, out_pred, eh_b, MROWS, EXPR_IN, D_MODEL, D_MODEL, 0);

  // --- flow matching head ---
  temb_kernel<<<BATCH, EW, 0, stream>>>(flow_t, ft1_W, ft1_b, ft2_W, ft2_b, temb);
  cat_kernel<<<(MROWS * (D_MODEL + EXPR_IN) + EW - 1) / EW, EW, 0, stream>>>(hid, temb, noisy, cat);
  gemm(cat, fv1_W, vtmp, fv1_b, MROWS, FLOW_HID, D_MODEL + EXPR_IN, D_MODEL + EXPR_IN, 0);
  ln_kernel<<<MROWS, EW, 0, stream>>>(vtmp, vtmp, fvln_g, fvln_b, FLOW_HID, /*gelu first*/1);
  gemm(vtmp, fv2_W, out_v, fv2_b, MROWS, EXPR_IN, FLOW_HID, FLOW_HID, 0);
}
